// MixedScoreMultiHeadAttention_54528904790831
// MI455X (gfx1250) — compile-verified, matrix operand order checked by reading, not run on hardware
//
#include <hip/hip_runtime.h>
#include <hip/hip_bf16.h>

// MixedScoreMultiHeadAttention for MI455X (gfx1250, wave32, WMMA bf16 path)

typedef __attribute__((ext_vector_type(16))) __bf16 v16bf;
typedef __attribute__((ext_vector_type(8)))  float  v8f;

#define HEADS 16
#define DQKV  16
#define EMBED 256
#define RC    512

__device__ __forceinline__ unsigned short f2bf(float f) {
  unsigned int u = __float_as_uint(f);
  u += 0x7FFFu + ((u >> 16) & 1u);   // round-to-nearest-even to bf16
  return (unsigned short)(u >> 16);
}

// Build a 16-element bf16 fragment from two 16B register quads (no per-element moves)
__device__ __forceinline__ v16bf make_frag(uint4 lo, uint4 hi) {
  union { uint4 v[2]; v16bf b; } u;
  u.v[0] = lo; u.v[1] = hi;
  return u.b;
}
__device__ __forceinline__ uint4 zero4() { uint4 z; z.x = z.y = z.z = z.w = 0u; return z; }

// ---------------- f32 -> bf16 conversion ----------------
__global__ void cvt_bf16(const float* __restrict__ in, unsigned short* __restrict__ out, int n) {
  int i = blockIdx.x * blockDim.x + threadIdx.x;
  if (i < n) out[i] = f2bf(in[i]);
}

// f32 [K=256][N=256] -> bf16 transposed [N][K] (B-fragment friendly: contiguous K per column)
__global__ void cvt_bf16_t(const float* __restrict__ in, unsigned short* __restrict__ out) {
  int i = blockIdx.x * blockDim.x + threadIdx.x;   // 0..65535
  int k = i >> 8, n = i & 255;
  out[n * EMBED + k] = f2bf(in[k * EMBED + n]);
}

// ---------------- QKV projection: X(1024x256) @ W(256x256) -> bf16 ----------------
// one wave per 16x16 output tile; col tile == head. Q/K head-major [b][h][s][d]; V d-major [b][h][d][s]
__global__ void qkv_gemm(const unsigned short* __restrict__ xrow,
                         const unsigned short* __restrict__ xcol,
                         const unsigned short* __restrict__ wqT,
                         const unsigned short* __restrict__ wkT,
                         const unsigned short* __restrict__ wvT,
                         unsigned short* __restrict__ qb,
                         unsigned short* __restrict__ kb,
                         unsigned short* __restrict__ vt) {
  const int lane = threadIdx.x & 31, wid = threadIdx.x >> 5;
  const int half = lane >> 4, lm = lane & 15;
  const int rt = blockIdx.x * 4 + wid;        // 0..63 row tiles (b*r = 1024 rows)
  const int h  = blockIdx.y;                  // head == 16-col tile
  const int z  = blockIdx.z;                  // 0=Q (row_emb), 1=K, 2=V (col_emb)
  const unsigned short* x  = (z == 0) ? xrow : xcol;
  const unsigned short* wT = (z == 0) ? wqT : ((z == 1) ? wkT : wvT);

  v8f acc = {};
  const unsigned short* xr = x + (rt * 16 + lm) * EMBED;        // A row for this lane
  const unsigned short* wr = wT + (h * 16 + lm) * EMBED;        // B column (transposed row)
  for (int k0 = 0; k0 < EMBED; k0 += 32) {
    uint4 a0 = *(const uint4*)(xr + k0 + half * 8);
    uint4 a1 = *(const uint4*)(xr + k0 + 16 + half * 8);
    uint4 b0 = *(const uint4*)(wr + k0 + half * 8);
    uint4 b1 = *(const uint4*)(wr + k0 + 16 + half * 8);
    acc = __builtin_amdgcn_wmma_f32_16x16x32_bf16(false, make_frag(a0, a1),
                                                  false, make_frag(b0, b1),
                                                  (short)0, acc, false, false);
  }
#pragma unroll
  for (int i = 0; i < 8; ++i) {
    int g = rt * 16 + i + 8 * half;
    int bb = g >> 9, s = g & (RC - 1);
    unsigned short v = f2bf(acc[i]);
    if (z == 2) vt[(((bb * HEADS) + h) * DQKV + lm) * RC + s] = v;            // [b][h][d][c]
    else {
      unsigned short* dst = (z == 0) ? qb : kb;
      dst[(((bb * HEADS) + h) * RC + s) * DQKV + lm] = v;                     // [b][h][s][d]
    }
  }
}

// ---------------- fused attention: QK^T -> MLP mix -> masked online softmax -> PV ----------------
// one wave per (b, h, 16-row tile); iterates 16 pairs of 16-column tiles (full-K PV wmma)
__global__ void attn_kernel(const unsigned short* __restrict__ qb,
                            const unsigned short* __restrict__ kb,
                            const unsigned short* __restrict__ vt,
                            const float* __restrict__ cost,
                            const unsigned char* __restrict__ mask,
                            const float* __restrict__ W1,   // [H][2][16]
                            const float* __restrict__ b1,   // [H][16]
                            const float* __restrict__ W2,   // [H][16]
                            const float* __restrict__ b2,   // [H]
                            unsigned short* __restrict__ oh) { // bf16 [b*r][256]
  __shared__ __align__(16) unsigned short ptile[4][16][40];   // 16x32 P tile, padded stride
  const int lane = threadIdx.x & 31, wid = threadIdx.x >> 5;
  const int half = lane >> 4, lm = lane & 15;
  const int rt = blockIdx.x * 4 + wid;   // 0..31 row tiles within batch
  const int h  = blockIdx.y;
  const int bb = blockIdx.z;

  // per-head MLP params in registers (broadcast loads through cache)
  float w1a[16], w1b[16], b1r[16], w2r[16];
#pragma unroll
  for (int m = 0; m < 16; ++m) {
    w1a[m] = W1[(h * 2 + 0) * 16 + m];
    w1b[m] = W1[(h * 2 + 1) * 16 + m];
    b1r[m] = b1[h * 16 + m];
    w2r[m] = W2[h * 16 + m];
  }
  const float b2r = b2[h];

  // mixed-score MLP applied to one (logit, cost) pair
  auto mlp = [&](float logit, float cv) -> float {
    float acc2 = b2r;
#pragma unroll
    for (int m = 0; m < 16; ++m) {
      float hid = fmaxf(logit * w1a[m] + cv * w1b[m] + b1r[m], 0.f);
      acc2 += hid * w2r[m];
    }
    return acc2;
  };

  // Q A-fragment (K padded 16 -> 32 with zeros)
  const unsigned short* qrow = qb + (((bb * HEADS) + h) * RC + rt * 16 + lm) * DQKV;
  const v16bf qa = make_frag(*(const uint4*)(qrow + half * 8), zero4());

  const unsigned short* kbase = kb + ((bb * HEADS) + h) * RC * DQKV;
  const unsigned short* vrow  = vt + (((bb * HEADS) + h) * DQKV + lm) * RC;  // this lane's d row

  v8f o = {};
  float mrow[8], lrow[8];
#pragma unroll
  for (int i = 0; i < 8; ++i) { mrow[i] = -3.4028235e38f; lrow[i] = 0.f; }

  for (int ct2 = 0; ct2 < RC / 32; ++ct2) {
    const int c0 = ct2 * 32;
    // two K B-fragments (each padded K), two S sub-tiles
    const v16bf kf0 = make_frag(*(const uint4*)(kbase + (c0 + lm) * DQKV + half * 8), zero4());
    const v16bf kf1 = make_frag(*(const uint4*)(kbase + (c0 + 16 + lm) * DQKV + half * 8), zero4());
    v8f z = {};
    v8f s0 = __builtin_amdgcn_wmma_f32_16x16x32_bf16(false, qa, false, kf0, (short)0, z, false, false);
    v8f s1 = __builtin_amdgcn_wmma_f32_16x16x32_bf16(false, qa, false, kf1, (short)0, z, false, false);

#pragma unroll
    for (int i = 0; i < 8; ++i) {
      const int row  = rt * 16 + i + 8 * half;
      const int idx0 = (bb * RC + row) * RC + c0 + lm;
      const int idx1 = idx0 + 16;
      float mx0 = mlp(s0[i] * 0.25f, cost[idx0]);    // 1/sqrt(16)
      float mx1 = mlp(s1[i] * 0.25f, cost[idx1]);
      if (!mask[idx0]) mx0 = -3.4028235e38f;
      if (!mask[idx1]) mx1 = -3.4028235e38f;

      // joint online softmax over both 16-col sub-tiles (16-lane half-group reduction)
      float tmax = fmaxf(mx0, mx1);
#pragma unroll
      for (int off = 8; off >= 1; off >>= 1) tmax = fmaxf(tmax, __shfl_xor(tmax, off, 32));
      float mnew = fmaxf(mrow[i], tmax);
      float corr = __expf(mrow[i] - mnew);
      float p0 = (mx0 <= -3.0e38f) ? 0.f : __expf(mx0 - mnew);
      float p1 = (mx1 <= -3.0e38f) ? 0.f : __expf(mx1 - mnew);
      float tsum = p0 + p1;
#pragma unroll
      for (int off = 8; off >= 1; off >>= 1) tsum += __shfl_xor(tsum, off, 32);
      lrow[i] = lrow[i] * corr + tsum;
      mrow[i] = mnew;
      o[i] *= corr;
      ptile[wid][i + 8 * half][lm]      = f2bf(p0);  // C-layout -> row-major 16x32 LDS tile
      ptile[wid][i + 8 * half][16 + lm] = f2bf(p1);
    }
    asm volatile("s_wait_dscnt 0" ::: "memory");     // wave-private LDS RAW fence

    // full-K P A-fragment (two b128 LDS reads) and V B-fragment (two b128 global reads)
    const unsigned short* prow = &ptile[wid][lm][0];
    const v16bf pa = make_frag(*(const uint4*)(prow + half * 8),
                               *(const uint4*)(prow + 16 + half * 8));
    const v16bf vf = make_frag(*(const uint4*)(vrow + c0 + half * 8),
                               *(const uint4*)(vrow + c0 + 16 + half * 8));
    o = __builtin_amdgcn_wmma_f32_16x16x32_bf16(false, pa, false, vf, (short)0, o, false, false);
  }

  // normalize and store bf16 head-concat output [b][r][h*16 + d]
#pragma unroll
  for (int i = 0; i < 8; ++i) {
    int row = rt * 16 + i + 8 * half;
    oh[(bb * RC + row) * EMBED + h * DQKV + lm] = f2bf(o[i] / lrow[i]);
  }
}

// ---------------- output projection: OH(1024x256 bf16) @ Wout -> f32 out ----------------
__global__ void out_gemm(const unsigned short* __restrict__ oh,
                         const unsigned short* __restrict__ woT,
                         float* __restrict__ out) {
  const int lane = threadIdx.x & 31, wid = threadIdx.x >> 5;
  const int half = lane >> 4, lm = lane & 15;
  const int rt = blockIdx.x * 4 + wid;   // 0..63
  const int ctile = blockIdx.y;          // 0..15
  v8f acc = {};
  const unsigned short* xr = oh + (rt * 16 + lm) * EMBED;
  const unsigned short* wr = woT + (ctile * 16 + lm) * EMBED;
  for (int k0 = 0; k0 < EMBED; k0 += 32) {
    uint4 a0 = *(const uint4*)(xr + k0 + half * 8);
    uint4 a1 = *(const uint4*)(xr + k0 + 16 + half * 8);
    uint4 b0 = *(const uint4*)(wr + k0 + half * 8);
    uint4 b1 = *(const uint4*)(wr + k0 + 16 + half * 8);
    acc = __builtin_amdgcn_wmma_f32_16x16x32_bf16(false, make_frag(a0, a1),
                                                  false, make_frag(b0, b1),
                                                  (short)0, acc, false, false);
  }
#pragma unroll
  for (int i = 0; i < 8; ++i) {
    int g = rt * 16 + i + 8 * half;
    out[g * EMBED + ctile * 16 + lm] = acc[i];
  }
}

extern "C" void kernel_launch(void* const* d_in, const int* in_sizes, int n_in,
                              void* d_out, int out_size, void* d_ws, size_t ws_size,
                              hipStream_t stream) {
  const float* row_emb = (const float*)d_in[0];
  const float* col_emb = (const float*)d_in[1];
  const float* cost    = (const float*)d_in[2];
  const unsigned char* mask = (const unsigned char*)d_in[3];
  const float* Wq   = (const float*)d_in[4];
  const float* Wk   = (const float*)d_in[5];
  const float* Wv   = (const float*)d_in[6];
  const float* Wout = (const float*)d_in[7];
  const float* W1   = (const float*)d_in[8];
  const float* b1   = (const float*)d_in[9];
  const float* W2   = (const float*)d_in[10];
  const float* b2   = (const float*)d_in[11];
  float* out = (float*)d_out;

  char* ws = (char*)d_ws;
  unsigned short* xrow = (unsigned short*)(ws + 0);        // 1024*256 bf16
  unsigned short* xcol = (unsigned short*)(ws + 524288);
  unsigned short* wqT  = (unsigned short*)(ws + 1048576);  // transposed 256*256 bf16 each
  unsigned short* wkT  = (unsigned short*)(ws + 1179648);
  unsigned short* wvT  = (unsigned short*)(ws + 1310720);
  unsigned short* woT  = (unsigned short*)(ws + 1441792);
  unsigned short* qb   = (unsigned short*)(ws + 1572864);  // [2][16][512][16] bf16
  unsigned short* kbuf = (unsigned short*)(ws + 2097152);  // [2][16][512][16] bf16
  unsigned short* vtb  = (unsigned short*)(ws + 2621440);  // [2][16][16][512] bf16 (d-major)
  unsigned short* oh   = (unsigned short*)(ws + 3145728);  // 1024*256 bf16

  cvt_bf16<<<262144 / 256, 256, 0, stream>>>(row_emb, xrow, 262144);
  cvt_bf16<<<262144 / 256, 256, 0, stream>>>(col_emb, xcol, 262144);
  cvt_bf16_t<<<65536 / 256, 256, 0, stream>>>(Wq,   wqT);
  cvt_bf16_t<<<65536 / 256, 256, 0, stream>>>(Wk,   wkT);
  cvt_bf16_t<<<65536 / 256, 256, 0, stream>>>(Wv,   wvT);
  cvt_bf16_t<<<65536 / 256, 256, 0, stream>>>(Wout, woT);

  qkv_gemm<<<dim3(16, 16, 3), 128, 0, stream>>>(xrow, xcol, wqT, wkT, wvT, qb, kbuf, vtb);
  attn_kernel<<<dim3(8, 16, 2), 128, 0, stream>>>(qb, kbuf, vtb, cost, mask,
                                                  W1, b1, W2, b2, oh);
  out_gemm<<<dim3(16, 16), 128, 0, stream>>>(oh, woT, out);
}
